// CayleyConvLanczos_63668595195981
// MI455X (gfx1250) — compile-verified, matrix operand order checked by reading, not run on hardware
//
#include <hip/hip_runtime.h>
#include <hip/hip_bf16.h>

// ---------------------------------------------------------------------------
// CayleyConvLanczos on MI455X (gfx1250), wave32.
//   v_wmma_f32_16x16x32_bf16 for all GEMMs, TDM (tensor_load_to_lds) staging
//   for the tall reduction GEMM P = eig_vecs^T @ x.
//
//   P  = eig_vecs^T @ x                     (256 x 256, reduce over 100000)
//   Q[k,o] = (P c0^T)[k,o] + 2*sum_j pr[j,k]*(P cr_j^T)[k,o]
//                          - 2*sum_j pi[j,k]*(P ci_j^T)[k,o]
//   out = eig_vecs @ Q                      (100000 x 256)
//
// Workspace (~6.84 MB):
//   [0)        Ppart : 25 * 256*256 f32 partials of P
//   [PBF_OFF)  Pbf   : 256*256 bf16
//   [QT_OFF)   Qt    : 256*256 bf16 (Q transposed: [o][k])
//   [PR_OFF)   pr    : 8*256 f32
//   [PI_OFF)   pi    : 8*256 f32
// ---------------------------------------------------------------------------

typedef __bf16 v16bf __attribute__((ext_vector_type(16)));
typedef float  v8f   __attribute__((ext_vector_type(8)));
typedef unsigned int u32x4 __attribute__((ext_vector_type(4)));
typedef int          i32x4 __attribute__((ext_vector_type(4)));
typedef int          i32x8 __attribute__((ext_vector_type(8)));

#define NROWS      100000
#define CDIM       256
#define NSPLIT     25
#define CHUNKS     3125          // NROWS / 32
#define CPS        125           // CHUNKS / NSPLIT
#define PPART_ELEMS (NSPLIT * CDIM * CDIM)
#define PBF_OFF    (size_t)(PPART_ELEMS * 4)
#define QT_OFF     (PBF_OFF + (size_t)(CDIM * CDIM * 2))
#define PR_OFF     (QT_OFF + (size_t)(CDIM * CDIM * 2))
#define PI_OFF     (PR_OFF + (size_t)(8 * CDIM * 4))

union ABuf {
  v16bf  v;
  __bf16 e[16];
  uint4  q[2];
};

// native f32 -> bf16 (RNE); lowers to v_cvt_pk_bf16_f32-class codegen
__device__ __forceinline__ __bf16 tobf(float f) { return (__bf16)f; }

__device__ __forceinline__ void cvt8(const float* __restrict__ p, __bf16* d) {
  float4 a = *(const float4*)p;
  float4 b = *(const float4*)(p + 4);
  d[0] = tobf(a.x); d[1] = tobf(a.y); d[2] = tobf(a.z); d[3] = tobf(a.w);
  d[4] = tobf(b.x); d[5] = tobf(b.y); d[6] = tobf(b.z); d[7] = tobf(b.w);
}

__device__ __forceinline__ v8f wmma_bf16(const ABuf& a, const ABuf& b, v8f c) {
  return __builtin_amdgcn_wmma_f32_16x16x32_bf16(
      false, a.v, false, b.v, (short)0, c, false, false);
}

// ---------------------------------------------------------------------------
// Tensor Data Mover: 2D tile (data_size = 4B) Global -> LDS.
// D# packing per CDNA5 ISA §8.3 (group0) / §8.4 (group1); groups 2/3 unused
// (2D tensor), passed zero-filled.  This toolchain exposes the 6-arg builtin:
//   (uint32x4 g0, int32x8 g1, int32x4 g2, int32x4 g3, int32x8 g4, i32 cpol)
// ---------------------------------------------------------------------------
#if defined(__has_builtin)
#if __has_builtin(__builtin_amdgcn_tensor_load_to_lds) && __has_builtin(__builtin_amdgcn_s_wait_tensorcnt)
#define USE_TDM 1
#endif
#endif

#ifdef USE_TDM
__device__ __forceinline__ void tdm_load_2d_f32(unsigned lds_addr, const void* gaddr,
                                                unsigned tile_w, unsigned tile_h,
                                                unsigned stride_elems,
                                                unsigned tensor_w, unsigned tensor_h) {
  unsigned long long ga = (unsigned long long)(uintptr_t)gaddr;
  u32x4 g0;
  g0[0] = 1u;                                                  // count=1 (valid user D#)
  g0[1] = lds_addr;                                            // lds_addr (bytes)
  g0[2] = (unsigned)(ga & 0xFFFFFFFFu);                        // global_addr[31:0]
  g0[3] = (unsigned)((ga >> 32) & 0x01FFFFFFu) | (2u << 30);   // addr[56:32] | type=2
  i32x8 g1;
  g1[0] = (int)(2u << 16);                                     // wg_mask=0, data_size=2 (4B)
  g1[1] = (int)(tensor_w << 16);                               // tensor_dim0[15:0] @63:48
  g1[2] = (int)((tensor_w >> 16) | (tensor_h << 16));          // dim0[31:16] | dim1[15:0]
  g1[3] = (int)((tensor_h >> 16) | (tile_w << 16));            // dim1[31:16] | tile_dim0
  g1[4] = (int)(tile_h & 0xFFFFu);                             // tile_dim1 (tile_dim2=0)
  g1[5] = (int)stride_elems;                                   // tensor_dim0_stride[31:0]
  g1[6] = 0;                                                   // stride hi / dim1_stride lo
  g1[7] = 0;
  i32x4 gz4 = {0, 0, 0, 0};
  i32x8 gz8 = {0, 0, 0, 0, 0, 0, 0, 0};
  __builtin_amdgcn_tensor_load_to_lds(g0, g1, gz4, gz4, gz8, 0);
}
#endif

// ---------------------------------------------------------------------------
// Kernel 1: P partials.  grid = (16 k-tiles, 25 splits), block = 512 (16 waves).
// Per 32-row chunk: TDM stages X[32 x 256] (32KB) and V[32 x 16] (2KB) into
// LDS; wave w computes the 16x16 (k-tile, i-tile=w) WMMA update from LDS.
// Fragment layout per ISA 7.12.2: lane L -> row/col L%16; element t ->
//   K(t) = (t&7) + 16*(t>>3) + 8*(L>>4)
// ---------------------------------------------------------------------------
__global__ void __launch_bounds__(512)
k1_p_partials(const float* __restrict__ V, const float* __restrict__ X,
              float* __restrict__ Ppart) {
  __shared__ float xs[32 * CDIM];   // 32 KB
  __shared__ float vs[32 * 16];     // 2 KB
  const int ktile = blockIdx.x;
  const int split = blockIdx.y;
  const int wave  = threadIdx.x >> 5;
  const int lane  = threadIdx.x & 31;
  const int grp   = lane >> 4;
  const int r     = lane & 15;
  const int kbase = ktile * 16;
  const int ibase = wave * 16;

  v8f acc = {};
  const int c0 = split * CPS, c1 = c0 + CPS;
  for (int c = c0; c < c1; ++c) {
    const int n0 = c * 32;
#ifdef USE_TDM
    if (wave == 0) {   // wave-uniform; TDM ignores EXEC, descriptor is scalar
      tdm_load_2d_f32((unsigned)(uintptr_t)xs, X + (size_t)n0 * CDIM,
                      /*tile_w=*/CDIM, /*tile_h=*/32, /*stride=*/CDIM,
                      /*tensor_w=*/CDIM, /*tensor_h=*/32);
      tdm_load_2d_f32((unsigned)(uintptr_t)vs, V + (size_t)n0 * CDIM + kbase,
                      /*tile_w=*/16, /*tile_h=*/32, /*stride=*/CDIM,
                      /*tensor_w=*/16, /*tensor_h=*/32);
      __builtin_amdgcn_s_wait_tensorcnt(0);
    }
    __syncthreads();   // LDS tiles visible to all 16 waves
    ABuf A, B;
#pragma unroll
    for (int t = 0; t < 16; ++t) {
      const int kk = (t & 7) + ((t >> 3) << 4) + (grp << 3);
      A.e[t] = tobf(vs[kk * 16 + r]);
      B.e[t] = tobf(xs[kk * CDIM + ibase + r]);
    }
    acc = wmma_bf16(A, B, acc);
    __syncthreads();   // all reads done before next chunk's DMA overwrites LDS
#else
    ABuf A, B;
#pragma unroll
    for (int t = 0; t < 16; ++t) {
      const int kk  = (t & 7) + ((t >> 3) << 4) + (grp << 3);
      const int row = (n0 + kk) * CDIM;
      A.e[t] = tobf(V[row + kbase + r]);
      B.e[t] = tobf(X[row + ibase + r]);
    }
    acc = wmma_bf16(A, B, acc);
#endif
  }

  float* out = Ppart + (size_t)split * (CDIM * CDIM);
#pragma unroll
  for (int v = 0; v < 8; ++v) {
    const int m = (grp << 3) + v;               // C/D layout: M = 8*grp + v
    out[(kbase + m) * CDIM + ibase + r] = acc[v];
  }
}

// ---------------------------------------------------------------------------
// Kernel 2: Cayley base powers. pr/pi[j][k] = Re/Im( base_k^(j+1) ).
// ---------------------------------------------------------------------------
__global__ void k2_powers(const float* __restrict__ eig, const float* __restrict__ hP,
                          const float* __restrict__ aP,
                          float* __restrict__ pr, float* __restrict__ pi) {
  const int k = blockIdx.x * blockDim.x + threadIdx.x;
  if (k >= CDIM) return;
  const float t  = hP[0] * (eig[k] - aP[0]);
  const float d  = t * t + 1.0f;
  const float br = (t * t - 1.0f) / d;   // (t-i)/(t+i) = (t^2-1 - 2ti)/(t^2+1)
  const float bi = (-2.0f * t) / d;
  float cr = br, ci = bi;
#pragma unroll
  for (int j = 0; j < 8; ++j) {
    pr[j * CDIM + k] = cr;
    pi[j * CDIM + k] = ci;
    const float nr = cr * br - ci * bi;
    ci = cr * bi + ci * br;
    cr = nr;
  }
}

// ---------------------------------------------------------------------------
// Kernel 3a: reduce P partials -> bf16 P. 256 blocks x 256 threads.
// ---------------------------------------------------------------------------
__global__ void k3a_reduce(const float* __restrict__ Ppart, unsigned short* __restrict__ Pbf) {
  const int idx = blockIdx.x * 256 + threadIdx.x;   // 65536 elements
  float s = 0.0f;
#pragma unroll
  for (int p = 0; p < NSPLIT; ++p) s += Ppart[(size_t)p * (CDIM * CDIM) + idx];
  __bf16 b = tobf(s);
  Pbf[idx] = *(unsigned short*)&b;
}

// ---------------------------------------------------------------------------
// Kernel 3b: Q. 256 waves (32 blocks x 8 waves), one 16x16 (k,o) tile each.
// 17 GEMM accumulations of P(16x256) x W^T(256x16) per tile; temp accumulators
// scaled per-row by 2*pr / -2*pi (row M = 8*grp+v in C/D layout).
// Writes Qt[o][k] in bf16 (16B-aligned runs of 8 along k).
// ---------------------------------------------------------------------------
__global__ void __launch_bounds__(256)
k3b_q(const unsigned short* __restrict__ Pbf, const float* __restrict__ c0w,
      const float* __restrict__ crw, const float* __restrict__ ciw,
      const float* __restrict__ pr, const float* __restrict__ pi,
      unsigned short* __restrict__ Qt) {
  const int wid   = (blockIdx.x * blockDim.x + threadIdx.x) >> 5;  // 0..255
  const int lane  = threadIdx.x & 31;
  const int grp   = lane >> 4;
  const int r     = lane & 15;
  const int ktile = wid >> 4;
  const int otile = wid & 15;
  const int kbase = ktile * 16;
  const int obase = otile * 16;

  const unsigned short* arow = Pbf + (kbase + r) * CDIM;  // A: lane row = k
  const int wrow = (obase + r) * CDIM;                    // B: lane col = o

  v8f f = {};

  // --- c0 term ---
  {
    v8f acc = {};
#pragma unroll
    for (int ic = 0; ic < 8; ++ic) {
      ABuf A, B;
      const int off = ic * 32 + (grp << 3);
      A.q[0] = *(const uint4*)(arow + off);
      A.q[1] = *(const uint4*)(arow + off + 16);
      cvt8(c0w + wrow + off, &B.e[0]);
      cvt8(c0w + wrow + off + 16, &B.e[8]);
      acc = wmma_bf16(A, B, acc);
    }
#pragma unroll
    for (int v = 0; v < 8; ++v) f[v] += acc[v];
  }

  // --- 2*Re( sum_j c_j * base^j ) terms ---
  for (int j = 0; j < 8; ++j) {
    const float* wr = crw + (size_t)j * (CDIM * CDIM);
    const float* wi = ciw + (size_t)j * (CDIM * CDIM);
    v8f aR = {}, aI = {};
#pragma unroll
    for (int ic = 0; ic < 8; ++ic) {
      ABuf A, BR, BI;
      const int off = ic * 32 + (grp << 3);
      A.q[0] = *(const uint4*)(arow + off);
      A.q[1] = *(const uint4*)(arow + off + 16);
      cvt8(wr + wrow + off, &BR.e[0]);
      cvt8(wr + wrow + off + 16, &BR.e[8]);
      cvt8(wi + wrow + off, &BI.e[0]);
      cvt8(wi + wrow + off + 16, &BI.e[8]);
      aR = wmma_bf16(A, BR, aR);
      aI = wmma_bf16(A, BI, aI);
    }
    const float4 p0 = *(const float4*)(pr + j * CDIM + kbase + (grp << 3));
    const float4 p1 = *(const float4*)(pr + j * CDIM + kbase + (grp << 3) + 4);
    const float4 q0 = *(const float4*)(pi + j * CDIM + kbase + (grp << 3));
    const float4 q1 = *(const float4*)(pi + j * CDIM + kbase + (grp << 3) + 4);
    const float prv[8] = {p0.x, p0.y, p0.z, p0.w, p1.x, p1.y, p1.z, p1.w};
    const float piv[8] = {q0.x, q0.y, q0.z, q0.w, q1.x, q1.y, q1.z, q1.w};
#pragma unroll
    for (int v = 0; v < 8; ++v)
      f[v] += 2.0f * (prv[v] * aR[v] - piv[v] * aI[v]);
  }

  // store Q transposed in bf16: Qt[obase+r][kbase + 8*grp + v]
  union { uint4 u; __bf16 e[8]; } st;
#pragma unroll
  for (int v = 0; v < 8; ++v) st.e[v] = tobf(f[v]);
  *(uint4*)(Qt + (obase + r) * CDIM + kbase + (grp << 3)) = st.u;
}

// ---------------------------------------------------------------------------
// Kernel 4: out = eig_vecs @ Q.  100000 waves (12500 blocks x 8 waves),
// one 16(m) x 16(o) output tile per wave, 8 WMMA k-steps of 32.
// ---------------------------------------------------------------------------
__global__ void __launch_bounds__(256)
k4_out(const float* __restrict__ V, const unsigned short* __restrict__ Qt,
       float* __restrict__ out) {
  const int wid   = (blockIdx.x * blockDim.x + threadIdx.x) >> 5;
  const int lane  = threadIdx.x & 31;
  const int grp   = lane >> 4;
  const int r     = lane & 15;
  const int mtile = wid >> 4;           // 0..6249 (exact: 6250*16 waves)
  const int otile = wid & 15;
  if (mtile >= NROWS / 16) return;      // wave-uniform; EXEC stays all-ones
  const int mbase = mtile * 16;
  const int obase = otile * 16;

  const float*          vrow = V  + (size_t)(mbase + r) * CDIM;  // A: lane row = m
  const unsigned short* qrow = Qt + (obase + r) * CDIM;          // B: lane col = o

  v8f acc = {};
#pragma unroll
  for (int kc = 0; kc < 8; ++kc) {
    ABuf A, B;
    const int off = kc * 32 + (grp << 3);
    cvt8(vrow + off, &A.e[0]);
    cvt8(vrow + off + 16, &A.e[8]);
    B.q[0] = *(const uint4*)(qrow + off);
    B.q[1] = *(const uint4*)(qrow + off + 16);
    acc = wmma_bf16(A, B, acc);
  }
#pragma unroll
  for (int v = 0; v < 8; ++v) {
    const int m = mbase + (grp << 3) + v;   // C/D layout: M = 8*grp + v
    out[(size_t)m * CDIM + obase + r] = acc[v];
  }
}

// ---------------------------------------------------------------------------
extern "C" void kernel_launch(void* const* d_in, const int* in_sizes, int n_in,
                              void* d_out, int out_size, void* d_ws, size_t ws_size,
                              hipStream_t stream) {
  const float* x    = (const float*)d_in[0];   // (N, 256)
  const float* eig  = (const float*)d_in[1];   // (256,)
  const float* V    = (const float*)d_in[2];   // (N, 256)
  const float* h    = (const float*)d_in[3];   // scalar
  const float* al   = (const float*)d_in[4];   // scalar
  const float* c0w  = (const float*)d_in[5];   // (256, 256)
  const float* crw  = (const float*)d_in[6];   // (8, 256, 256)
  const float* ciw  = (const float*)d_in[7];   // (8, 256, 256)
  float* out = (float*)d_out;

  char* ws = (char*)d_ws;                      // needs ~6.84 MB
  float*          Ppart = (float*)ws;
  unsigned short* Pbf   = (unsigned short*)(ws + PBF_OFF);
  unsigned short* Qt    = (unsigned short*)(ws + QT_OFF);
  float*          pr    = (float*)(ws + PR_OFF);
  float*          pi    = (float*)(ws + PI_OFF);

  k1_p_partials<<<dim3(16, NSPLIT), 512, 0, stream>>>(V, x, Ppart);
  k2_powers<<<1, 256, 0, stream>>>(eig, h, al, pr, pi);
  k3a_reduce<<<256, 256, 0, stream>>>(Ppart, Pbf);
  k3b_q<<<32, 256, 0, stream>>>(Pbf, c0w, crw, ciw, pr, pi, Qt);
  k4_out<<<12500, 256, 0, stream>>>(V, Qt, out);
}
